// GATModel_12463995093411
// MI455X (gfx1250) — matrix operand order repaired and checked by reading, not compiled, not run on hardware
//
#include <hip/hip_runtime.h>
#include <hip/hip_bf16.h>
#include <stdint.h>

#define H_HEADS   8
#define C_CH      32
#define EMBD      256
#define IN_DIMC   128
#define EDGE_DIMC 32
#define L_LAYERS  4
#define NEG_SLOPE 0.2f
#define LDS_PAD   8       // 8 halfs = 16B per row -> conflict-free ds_load_b128

typedef __attribute__((ext_vector_type(16))) __bf16    v16bf;
typedef __attribute__((ext_vector_type(8)))  float     v8f;
typedef __attribute__((ext_vector_type(4)))  unsigned  v4u;
typedef __attribute__((ext_vector_type(4)))  unsigned  u32x4;
typedef __attribute__((ext_vector_type(8)))  int       i32x8;
typedef __attribute__((ext_vector_type(4)))  int       i32x4;

union BF16x16 { v4u u[2]; v16bf v; };

// monotonic float<->uint encoding so atomicMax(u32) == float max
__device__ __forceinline__ unsigned fkey(float f) {
  unsigned b = __float_as_uint(f);
  return (b & 0x80000000u) ? ~b : (b | 0x80000000u);
}
__device__ __forceinline__ float fdec(unsigned k) {
  unsigned b = (k & 0x80000000u) ? (k & 0x7fffffffu) : ~k;
  return __uint_as_float(b);
}

// ---------------- utility fills / converts ----------------
__global__ void fill_f32(float* p, float v, size_t n) {
  size_t i = (size_t)blockIdx.x * blockDim.x + threadIdx.x;
  if (i < n) p[i] = v;
}
__global__ void fill_u32(unsigned* p, unsigned v, size_t n) {
  size_t i = (size_t)blockIdx.x * blockDim.x + threadIdx.x;
  if (i < n) p[i] = v;
}
__global__ void f32_to_bf16(const float* __restrict__ in, __hip_bfloat16* __restrict__ out, size_t n) {
  size_t i = (size_t)blockIdx.x * blockDim.x + threadIdx.x;
  if (i < n) out[i] = __float2bfloat16(in[i]);
}
// W [K, 256] row-major -> Bt [256, K] (col-major of W) in bf16
__global__ void transpose_to_bf16(const float* __restrict__ Wm, __hip_bfloat16* __restrict__ Bt, int K) {
  int idx = blockIdx.x * blockDim.x + threadIdx.x;
  if (idx >= K * EMBD) return;
  int k = idx >> 8;       // row in W
  int n = idx & 255;      // col in W
  Bt[(size_t)n * K + k] = __float2bfloat16(Wm[idx]);
}
// h_next[n,d] = bias[d]
__global__ void init_bias(float* __restrict__ hn, const float* __restrict__ bias, size_t n) {
  size_t i = (size_t)blockIdx.x * blockDim.x + threadIdx.x;
  if (i < n) hn[i] = bias[i & (EMBD - 1)];
}

// ---------------- mean of edge_attr over E edges ----------------
__global__ void ea_mean(const float* __restrict__ ea, float* __restrict__ meanv, int E) {
  __shared__ float sm[EDGE_DIMC];
  int j = threadIdx.x & 31;                 // stride keeps column constant
  if (threadIdx.x < EDGE_DIMC) sm[threadIdx.x] = 0.f;
  __syncthreads();
  float acc = 0.f;
  size_t total = (size_t)E * EDGE_DIMC;
  size_t stride = (size_t)gridDim.x * blockDim.x;   // multiple of 32
  for (size_t i = (size_t)blockIdx.x * blockDim.x + threadIdx.x; i < total; i += stride)
    acc += ea[i];
  atomicAdd(&sm[j], acc);
  __syncthreads();
  if (threadIdx.x < EDGE_DIMC)
    atomicAdd(&meanv[threadIdx.x], sm[threadIdx.x] * (1.0f / (float)E));
}

// ---------------- WMMA GEMM: C[nRows,256] = A[nRows,K](bf16) x B[K,256], Bt = B^T (bf16) ---------------
// Block = 8 waves = 128 rows x 64 cols. B tile (64 x K bf16, ~32KB) staged into
// LDS by the Tensor Data Mover (one descriptor, padded rows for bank-conflict-free
// ds_load_b128 fragment reads); A streams from global.
__global__ void __launch_bounds__(256)
gemm_bf16_wmma(const __hip_bfloat16* __restrict__ A,
               const __hip_bfloat16* __restrict__ Bt,
               const float* __restrict__ bias,      // may be null
               float* __restrict__ C,
               int nRows, int K) {
  __shared__ __hip_bfloat16 sB[64 * (EMBD + LDS_PAD)];   // 33 KB max
  const int KS   = K + LDS_PAD;                          // padded LDS row stride (halfs)
  const int col0 = blockIdx.y * 64;

#if defined(__gfx1250__) && __has_builtin(__builtin_amdgcn_tensor_load_to_lds)
  if (threadIdx.x == 0) {
    // D# group0: count=1 | lds_addr | global_addr(57b) | type=2
    unsigned ldsOff = (unsigned)(size_t)(void*)&sB[0];   // low 32b of flat LDS addr = LDS offset
    uint64_t ga = (uint64_t)(const void*)(Bt + (size_t)col0 * K);
    u32x4 g0 = { 1u, ldsOff, (unsigned)ga,
                 ((unsigned)((ga >> 32) & 0x01FFFFFFu)) | 0x80000000u };
    // D# group1: data_size=2B, pad every row (K/2 DWORDs) by 4 DWORDs (16B),
    // tensor_dim0 = K, tensor_dim1 = 64, tile = K x 64, dim0_stride = K.
    int padcode = (K == 256) ? 6 : 5;                    // 2^(c+1) DW = K/2 DW
    i32x8 g1 = { (int)((1u << 16) | (1u << 20) | ((unsigned)padcode << 22) | (3u << 25)),
                 (int)((unsigned)K << 16),               // tensor_dim0 lo16 in [63:48]
                 (int)(64u << 16),                       // tensor_dim1 lo16 in [95:80]
                 (int)((unsigned)K << 16),               // tile_dim0 in [127:112]
                 64,                                     // tile_dim1 in [143:128]
                 K,                                      // tensor_dim0_stride lo32
                 0, 0 };
    i32x4 gz = { 0, 0, 0, 0 };
#if __clang_major__ >= 23
    i32x8 gz8 = { 0, 0, 0, 0, 0, 0, 0, 0 };
    __builtin_amdgcn_tensor_load_to_lds(g0, g1, gz, gz, gz8, 0);
#else
    __builtin_amdgcn_tensor_load_to_lds(g0, g1, gz, gz, 0);
#endif
    __builtin_amdgcn_s_wait_tensorcnt(0);
  }
  __syncthreads();
#else
  // cooperative staging fallback (also taken on host compile pass)
  const int chunksPerRow = K >> 3;
  for (int i = threadIdx.x; i < 64 * chunksPerRow; i += blockDim.x) {
    int r = i / chunksPerRow, c = i - r * chunksPerRow;
    *reinterpret_cast<v4u*>(&sB[r * KS + c * 8]) =
        *reinterpret_cast<const v4u*>(Bt + (size_t)(col0 + r) * K + c * 8);
  }
  __syncthreads();
#endif

  const int lane = threadIdx.x & 31;
  const int wave = threadIdx.x >> 5;
  const int row0 = blockIdx.x * 128 + wave * 16;
  if (row0 >= nRows) return;                 // after the barrier
  const int m  = lane & 15;
  const int hi = lane >> 4;                  // 0: lanes 0-15, 1: lanes 16-31

  int ar = row0 + m; if (ar >= nRows) ar = nRows - 1;       // clamp (stores guarded)
  const __hip_bfloat16* Arow = A + (size_t)ar * K + hi * 8; // A frag: K 0-7/16-23 (lo) or 8-15/24-31 (hi)
  __builtin_prefetch(Arow, 0, 1);
  const int kB = hi * 8;                                    // B frag: K 0-7/16-23 (lo) or 8-15/24-31 (hi)

  v8f z = {0.f,0.f,0.f,0.f,0.f,0.f,0.f,0.f};
  v8f acc[4]; acc[0] = z; acc[1] = z; acc[2] = z; acc[3] = z;

  for (int k0 = 0; k0 < K; k0 += 32) {
    BF16x16 a;
    a.u[0] = *reinterpret_cast<const v4u*>(Arow + k0);        // K = k0+hi*8 + 0..7
    a.u[1] = *reinterpret_cast<const v4u*>(Arow + k0 + 16);   // K = k0+hi*8 + 16..23
#pragma unroll
    for (int ct = 0; ct < 4; ++ct) {
      const __hip_bfloat16* bp = sB + (size_t)(ct * 16 + m) * KS + k0 + kB;
      BF16x16 b;
      b.u[0] = *reinterpret_cast<const v4u*>(bp);             // K = k0+kB + 0..7
      b.u[1] = *reinterpret_cast<const v4u*>(bp + 16);        // K = k0+kB + 16..23
      acc[ct] = __builtin_amdgcn_wmma_f32_16x16x32_bf16(
          false, a.v, false, b.v, (short)0, acc[ct], false, false);
    }
  }

  // C/D layout: VGPR v -> row (v + hi*8), col = lane&15
  const int rbase = row0 + hi * 8;
#pragma unroll
  for (int ct = 0; ct < 4; ++ct) {
    const int c = col0 + ct * 16 + m;
    const float bv = bias ? bias[c] : 0.f;
#pragma unroll
    for (int v = 0; v < 8; ++v) {
      int r = rbase + v;
      if (r < nRows) C[(size_t)r * EMBD + c] = acc[ct][v] + bv;
    }
  }
}

// ---------------- a_s / a_d: per-head dot of hp rows with att vectors ----------------
__global__ void attn_sd(const float* __restrict__ hp,
                        const float* __restrict__ att_src, const float* __restrict__ att_dst,
                        float* __restrict__ a_s, float* __restrict__ a_d, int N) {
  __shared__ float ss[EMBD], sd[EMBD];
  for (int i = threadIdx.x; i < EMBD; i += blockDim.x) { ss[i] = att_src[i]; sd[i] = att_dst[i]; }
  __syncthreads();
  int idx = blockIdx.x * blockDim.x + threadIdx.x;
  if (idx >= N * H_HEADS) return;
  int n = idx >> 3, h = idx & 7;
  const float* row = hp + (size_t)n * EMBD + h * C_CH;
  const float* s0 = ss + h * C_CH;
  const float* d0 = sd + h * C_CH;
  float as = 0.f, ad = 0.f;
#pragma unroll
  for (int c = 0; c < C_CH; c += 4) {
    float4 v = *reinterpret_cast<const float4*>(row + c);
    as = fmaf(v.x, s0[c], as); as = fmaf(v.y, s0[c+1], as);
    as = fmaf(v.z, s0[c+2], as); as = fmaf(v.w, s0[c+3], as);
    ad = fmaf(v.x, d0[c], ad); ad = fmaf(v.y, d0[c+1], ad);
    ad = fmaf(v.z, d0[c+2], ad); ad = fmaf(v.w, d0[c+3], ad);
  }
  a_s[idx] = as; a_d[idx] = ad;
}

// ---------------- fold att_edge into We_l: wea[d][h] = sum_c We[d, h*32+c]*att_edge[h,c] ----------------
__global__ void edge_att_proj(const float* __restrict__ We, const float* __restrict__ att_edge,
                              float* __restrict__ wea) {
  int t = threadIdx.x;          // 256 threads, one block
  int d = t >> 3, h = t & 7;
  const float* wrow = We + (size_t)d * EMBD + h * C_CH;
  const float* arow = att_edge + h * C_CH;
  float acc = 0.f;
#pragma unroll
  for (int c = 0; c < C_CH; ++c) acc = fmaf(wrow[c], arow[c], acc);
  wea[d * 8 + h] = acc;
}

// ---------------- edge logits + leaky-relu + segment max (atomic u32 max) ----------------
__global__ void edge_logits(const int* __restrict__ ei, const float* __restrict__ ea,
                            const float* __restrict__ meanv, const float* __restrict__ wea,
                            const float* __restrict__ a_s, const float* __restrict__ a_d,
                            float* __restrict__ ev, unsigned* __restrict__ mbuf,
                            int Etot, int E) {
  __shared__ float sw[EDGE_DIMC * H_HEADS];
  __shared__ float sm[EDGE_DIMC];
  sw[threadIdx.x] = wea[threadIdx.x];
  if (threadIdx.x < EDGE_DIMC) sm[threadIdx.x] = meanv[threadIdx.x];
  __syncthreads();
  int e = blockIdx.x * blockDim.x + threadIdx.x;
  if (e >= Etot) return;
  int s, d;
  float acc[8] = {0.f,0.f,0.f,0.f,0.f,0.f,0.f,0.f};
  if (e < E) {
    s = ei[e]; d = ei[E + e];
    const float* row = ea + (size_t)e * EDGE_DIMC;
#pragma unroll
    for (int j = 0; j < EDGE_DIMC; j += 4) {
      float4 v = *reinterpret_cast<const float4*>(row + j);
#pragma unroll
      for (int h = 0; h < 8; ++h) {
        acc[h] = fmaf(v.x, sw[(j    ) * 8 + h], acc[h]);
        acc[h] = fmaf(v.y, sw[(j + 1) * 8 + h], acc[h]);
        acc[h] = fmaf(v.z, sw[(j + 2) * 8 + h], acc[h]);
        acc[h] = fmaf(v.w, sw[(j + 3) * 8 + h], acc[h]);
      }
    }
  } else {
    s = d = e - E;                         // self loop, edge_attr = mean
#pragma unroll
    for (int j = 0; j < EDGE_DIMC; ++j) {
      float aj = sm[j];
#pragma unroll
      for (int h = 0; h < 8; ++h) acc[h] = fmaf(aj, sw[j * 8 + h], acc[h]);
    }
  }
  size_t sb = (size_t)s * 8, db = (size_t)d * 8, eb = (size_t)e * 8;
#pragma unroll
  for (int h = 0; h < 8; ++h) {
    float v = a_s[sb + h] + a_d[db + h] + acc[h];
    v = (v >= 0.f) ? v : NEG_SLOPE * v;
    ev[eb + h] = v;
    atomicMax(&mbuf[db + h], fkey(v));
  }
}

// ---------------- exp(e - max) + segment sum ----------------
__global__ void edge_exp(const int* __restrict__ ei, float* __restrict__ ev,
                         const unsigned* __restrict__ mbuf, float* __restrict__ denom,
                         int Etot, int E) {
  size_t idx = (size_t)blockIdx.x * blockDim.x + threadIdx.x;
  if (idx >= (size_t)Etot * 8) return;
  int e = (int)(idx >> 3), h = (int)(idx & 7);
  int d = (e < E) ? ei[E + e] : (e - E);
  float mx = fdec(mbuf[(size_t)d * 8 + h]);
  float x = __expf(ev[idx] - mx);
  ev[idx] = x;
  atomicAdd(&denom[(size_t)d * 8 + h], x);
}

// ---------------- message aggregation: one wave per edge, lane = channel ----------------
__global__ void aggregate(const int* __restrict__ ei, const float* __restrict__ hp,
                          const float* __restrict__ ev, const float* __restrict__ denom,
                          float* __restrict__ hn, int Etot, int E) {
  int gid = blockIdx.x * blockDim.x + threadIdx.x;
  int lane = gid & 31;
  int e = gid >> 5;
  if (e >= Etot) return;
  int s, d;
  if (e < E) { s = ei[e]; d = ei[E + e]; } else { s = d = e - E; }
  const float* hs = hp + (size_t)s * EMBD;
  float* hd = hn + (size_t)d * EMBD;
  size_t eb = (size_t)e * 8, db = (size_t)d * 8;
#pragma unroll
  for (int h = 0; h < 8; ++h) {
    float alpha = ev[eb + h] / (denom[db + h] + 1e-16f);
    atomicAdd(&hd[h * C_CH + lane], hs[h * C_CH + lane] * alpha);
  }
}

// ---------------- final projection: out[n] = dot(h[n,:], w_out) + b_out ----------------
__global__ void out_proj(const float* __restrict__ h, const float* __restrict__ w_out,
                         const float* __restrict__ b_out, float* __restrict__ out, int N) {
  int gid = blockIdx.x * blockDim.x + threadIdx.x;
  int lane = gid & 31;
  int n = gid >> 5;
  if (n >= N) return;
  const float* row = h + (size_t)n * EMBD;
  float acc = 0.f;
#pragma unroll
  for (int c = 0; c < EMBD / 32; ++c) acc = fmaf(row[c * 32 + lane], w_out[c * 32 + lane], acc);
#pragma unroll
  for (int off = 16; off > 0; off >>= 1) acc += __shfl_xor(acc, off, 32);
  if (lane == 0) out[n] = acc + b_out[0];
}

extern "C" void kernel_launch(void* const* d_in, const int* in_sizes, int n_in,
                              void* d_out, int out_size, void* d_ws, size_t ws_size,
                              hipStream_t stream) {
  const float* x        = (const float*)d_in[0];
  const int*   ei       = (const int*)d_in[1];
  const float* ea       = (const float*)d_in[2];
  const float* w_in     = (const float*)d_in[3];
  const float* b_in     = (const float*)d_in[4];
  const float* W_l      = (const float*)d_in[5];
  const float* We_l     = (const float*)d_in[6];
  const float* att_src  = (const float*)d_in[7];
  const float* att_dst  = (const float*)d_in[8];
  const float* att_edge = (const float*)d_in[9];
  const float* bias_l   = (const float*)d_in[10];
  const float* w_out    = (const float*)d_in[11];
  const float* b_out    = (const float*)d_in[12];
  float* out = (float*)d_out;

  const int N = in_sizes[0] / IN_DIMC;
  const int E = in_sizes[1] / 2;
  const int Etot = E + N;

  char* wp = (char*)d_ws;
  auto carve = [&](size_t bytes) { char* p = wp; wp += (bytes + 255) & ~(size_t)255; return p; };
  float*          hA    = (float*)carve((size_t)N * EMBD * 4);
  float*          hB    = (float*)carve((size_t)N * EMBD * 4);
  float*          hp    = (float*)carve((size_t)N * EMBD * 4);
  __hip_bfloat16* hbf   = (__hip_bfloat16*)carve((size_t)N * EMBD * 2);
  __hip_bfloat16* Bt    = (__hip_bfloat16*)carve((size_t)EMBD * EMBD * 2);
  float*          a_s   = (float*)carve((size_t)N * 8 * 4);
  float*          a_d   = (float*)carve((size_t)N * 8 * 4);
  float*          ev    = (float*)carve((size_t)Etot * 8 * 4);
  unsigned*       mbuf  = (unsigned*)carve((size_t)N * 8 * 4);
  float*          denom = (float*)carve((size_t)N * 8 * 4);
  float*          meanv = (float*)carve(EDGE_DIMC * 4);
  float*          wea   = (float*)carve(EDGE_DIMC * H_HEADS * 4);

  const int B = 256;
  auto blks = [](size_t n, int b) { return (int)((n + (size_t)b - 1) / b); };

  // mean edge_attr (self-loop fill value)
  fill_f32<<<1, 32, 0, stream>>>(meanv, 0.f, EDGE_DIMC);
  ea_mean<<<512, B, 0, stream>>>(ea, meanv, E);

  // h0 = x @ w_in + b_in
  f32_to_bf16<<<blks((size_t)N * IN_DIMC, B), B, 0, stream>>>(x, hbf, (size_t)N * IN_DIMC);
  transpose_to_bf16<<<blks((size_t)IN_DIMC * EMBD, B), B, 0, stream>>>(w_in, Bt, IN_DIMC);
  dim3 ggrid((N + 127) / 128, EMBD / 64);
  gemm_bf16_wmma<<<ggrid, B, 0, stream>>>(hbf, Bt, b_in, hA, N, IN_DIMC);

  float* hcur = hA;
  float* hnext = hB;
  for (int l = 0; l < L_LAYERS; ++l) {
    f32_to_bf16<<<blks((size_t)N * EMBD, B), B, 0, stream>>>(hcur, hbf, (size_t)N * EMBD);
    transpose_to_bf16<<<blks((size_t)EMBD * EMBD, B), B, 0, stream>>>(W_l + (size_t)l * EMBD * EMBD, Bt, EMBD);
    gemm_bf16_wmma<<<ggrid, B, 0, stream>>>(hbf, Bt, nullptr, hp, N, EMBD);

    attn_sd<<<blks((size_t)N * H_HEADS, B), B, 0, stream>>>(hp, att_src + l * EMBD, att_dst + l * EMBD, a_s, a_d, N);
    edge_att_proj<<<1, 256, 0, stream>>>(We_l + (size_t)l * EDGE_DIMC * EMBD, att_edge + l * EMBD, wea);

    fill_u32<<<blks((size_t)N * 8, B), B, 0, stream>>>(mbuf, 0u, (size_t)N * 8);
    fill_f32<<<blks((size_t)N * 8, B), B, 0, stream>>>(denom, 0.f, (size_t)N * 8);
    init_bias<<<blks((size_t)N * EMBD, B), B, 0, stream>>>(hnext, bias_l + l * EMBD, (size_t)N * EMBD);

    edge_logits<<<blks((size_t)Etot, B), B, 0, stream>>>(ei, ea, meanv, wea, a_s, a_d, ev, mbuf, Etot, E);
    edge_exp<<<blks((size_t)Etot * 8, B), B, 0, stream>>>(ei, ev, mbuf, denom, Etot, E);
    aggregate<<<blks((size_t)Etot * 32, B), B, 0, stream>>>(ei, hp, ev, denom, hnext, Etot, E);

    float* t = hcur; hcur = hnext; hnext = t;
  }

  out_proj<<<blks((size_t)N * 32, B), B, 0, stream>>>(hcur, w_out, b_out, out, N);
}
